// BasicBlock_37503654429268
// MI455X (gfx1250) — 8-bit matrix operand order repaired and checked by reading, compiled, not run on hardware
//
#include <hip/hip_runtime.h>

typedef __attribute__((ext_vector_type(8))) int v8i;

#define EPS_ 1e-5f
#define LP 272

#define SC_AMAX 0
#define SC_SACT 8
#define SC_SW1  11
#define SC_SW2  14
#define SC_INV1 17
#define SC_S2   20
#define SC_INV2 23
#define SC_BNB  32
#define SC_ALPHA 288

#define QW_EXP_INTS 147456
#define QA_EXP_BYTES (32ull*1024*256)

__global__ void k_amax(const float* __restrict__ x, int n, int useAbs,
                       unsigned* __restrict__ out) {
  float m = 0.f;
  for (int i = blockIdx.x * blockDim.x + threadIdx.x; i < n;
       i += gridDim.x * blockDim.x) {
    float v = x[i];
    v = useAbs ? fabsf(v) : fmaxf(v, 0.f);
    m = fmaxf(m, v);
  }
  for (int off = 16; off >= 1; off >>= 1) m = fmaxf(m, __shfl_xor(m, off, 32));
  if ((threadIdx.x & 31) == 0) atomicMax(out, __float_as_uint(m));
}

__global__ void k_derive1(float* __restrict__ scal, const float* __restrict__ g,
                          const float* __restrict__ be, const float* __restrict__ mu,
                          const float* __restrict__ var) {
  unsigned* u = (unsigned*)scal;
  int t = threadIdx.x;
  if (t < 3) {
    const int lvs[3] = {4, 16, 256};
    int lv = lvs[t];
    float ax = fmaxf(__uint_as_float(u[0]), 1e-8f);
    float sact = (float)(lv - 1) / ax;
    float nn = (float)(lv / 2 - 1);
    float sw1 = nn / fmaxf(__uint_as_float(u[1]), 1e-8f);
    float sw2 = nn / fmaxf(__uint_as_float(u[2]), 1e-8f);
    scal[SC_SACT + t] = sact;
    scal[SC_SW1 + t] = sw1;
    scal[SC_SW2 + t] = sw2;
    scal[SC_INV1 + t] = 1.f / (sact * sw1);
  }
  __syncthreads();
  if (t < 256) {
    float A = g[t] * rsqrtf(var[t] + EPS_);
    float Bc = be[t] - mu[t] * A;
    scal[SC_BNB + t] = Bc;
    for (int e = 0; e < 3; e++)
      scal[SC_ALPHA + e * 256 + t] = A * scal[SC_INV1 + e];
  }
}

__global__ void k_derive2(float* __restrict__ scal) {
  unsigned* u = (unsigned*)scal;
  int e = threadIdx.x;
  if (e < 3) {
    const int lvs[3] = {4, 16, 256};
    float ah = fmaxf(__uint_as_float(u[3 + e]), 1e-8f);
    float s2 = (float)(lvs[e] - 1) / ah;
    scal[SC_S2 + e] = s2;
    scal[SC_INV2 + e] = 1.f / (s2 * scal[SC_SW2 + e]);
  }
}

__global__ void k_pack_w(const float* __restrict__ w, int* __restrict__ dst,
                         const float* __restrict__ scal, int swBase) {
  int t = blockIdx.x * blockDim.x + threadIdx.x;
  if (t >= 3 * QW_EXP_INTS) return;
  int dw = t & 7;
  int lane = (t >> 3) & 31;
  int cot = (t >> 8) & 15;
  int cb = (t >> 12) & 3;
  int tc = t >> 14;
  int tap = tc % 9;
  int e = tc / 9;
  const int nvals[3] = {1, 7, 127};
  float s = scal[swBase + e];
  float nf = (float)nvals[e];
  int m = lane & 15;
  int co = cot * 16 + m;
  int ky = tap / 3, kx = tap % 3;
  int khalf = (lane >> 4) ? 16 : 0;
  unsigned pack = 0;
  for (int j = 0; j < 4; j++) {
    int k = ((dw & 3) * 4) + (((dw >> 2) & 1) * 32) + khalf + j;
    int ci = cb * 64 + k;
    float wv = w[((co * 256 + ci) * 3 + ky) * 3 + kx];
    int qi = (int)rintf(fminf(fmaxf(wv * s, -nf), nf));
    pack |= ((unsigned)(qi & 0xff)) << (8 * j);
  }
  dst[t] = (int)pack;
}

__global__ void k_quant_x(const float* __restrict__ x, unsigned char* __restrict__ qa1,
                          const float* __restrict__ scal) {
  int t = blockIdx.x * blockDim.x + threadIdx.x;
  int p = t & 1023;
  int c4 = (t >> 10) & 63;
  int b = (t >> 16) & 31;
  int e = t >> 21;
  if (e >= 3) return;
  float s = scal[SC_SACT + e];
  unsigned pack = 0;
  for (int j = 0; j < 4; j++) {
    float v = x[(size_t)(b * 256 + c4 * 4 + j) * 1024 + p];
    int q = (int)rintf(fmaxf(v, 0.f) * s);
    q = min(q, 255);
    pack |= ((unsigned)q) << (8 * j);
  }
  *(unsigned*)(qa1 + (((size_t)(e * 32 + b) * 1024 + p) * 256 + c4 * 4)) = pack;
}

template <int MODE>
__global__ __launch_bounds__(256) void k_conv(
    const unsigned char* __restrict__ qaBase, const int* __restrict__ qwBase,
    const float* __restrict__ scal, const int* __restrict__ mask,
    unsigned char* __restrict__ qa2, float* __restrict__ fout,
    unsigned* __restrict__ amaxH) {
  __shared__ unsigned char lds[6 * 34 * LP];
  int bx = blockIdx.x;
  int e, b, rowblk;
  if (MODE == 0) { e = bx >> 8; b = (bx >> 3) & 31; rowblk = bx & 7; }
  else           { b = bx >> 3; rowblk = bx & 7; e = mask[b]; }

  const unsigned char* qa =
      qaBase + ((MODE == 2) ? ((size_t)b * 1024 * 256)
                            : ((size_t)(e * 32 + b) * 1024 * 256));
  const int* qw = qwBase + (size_t)e * QW_EXP_INTS;
  int tid = threadIdx.x;
  int rowbase = rowblk * 4;

  for (int i = tid; i < 204 * 16; i += 256) {
    int posIdx = i >> 4;
    int chunk = i & 15;
    int r = posIdx / 34;
    int c = posIdx % 34;
    int gr = rowbase + r - 1;
    int gc = c - 1;
    int4 val = {0, 0, 0, 0};
    if (gr >= 0 && gr < 32 && gc >= 0 && gc < 32)
      val = *(const int4*)(qa + ((size_t)(gr * 32 + gc) * 256 + chunk * 16));
    *(int4*)(&lds[posIdx * LP + chunk * 16]) = val;
  }
  __syncthreads();

  int wave = tid >> 5;
  int lane = tid & 31;
  int n = lane & 15;
  int khalf16 = (lane >> 4) << 4;

  v8i acc[2][8];
  v8i zero = {};
  for (int i = 0; i < 2; i++)
    for (int j = 0; j < 8; j++) acc[i][j] = zero;

  int pr[8], pc[8];
  for (int pt = 0; pt < 8; pt++) {
    int pos = pt * 16 + n;
    pr[pt] = pos >> 5;
    pc[pt] = pos & 31;
  }

  for (int tap = 0; tap < 9; tap++) {
    int dy = tap / 3 - 1, dx = tap % 3 - 1;
    for (int cb = 0; cb < 4; cb++) {
      v8i aF0 = *(const v8i*)(qw + (size_t)((((tap * 4 + cb) * 16 + wave * 2 + 0) * 32 + lane) * 8));
      v8i aF1 = *(const v8i*)(qw + (size_t)((((tap * 4 + cb) * 16 + wave * 2 + 1) * 32 + lane) * 8));
      for (int pt = 0; pt < 8; pt++) {
        const unsigned char* bp =
            &lds[(size_t)(((pr[pt] + 1 + dy) * 34 + (pc[pt] + 1 + dx)) * LP) +
                 cb * 64 + khalf16];
        v8i bF;
        bF[0] = *(const int*)(bp + 0);
        bF[1] = *(const int*)(bp + 4);
        bF[2] = *(const int*)(bp + 8);
        bF[3] = *(const int*)(bp + 12);
        bF[4] = *(const int*)(bp + 32);
        bF[5] = *(const int*)(bp + 36);
        bF[6] = *(const int*)(bp + 40);
        bF[7] = *(const int*)(bp + 44);
        acc[0][pt] = __builtin_amdgcn_wmma_i32_16x16x64_iu8(true, aF0, false, bF, acc[0][pt], false, false);
        acc[1][pt] = __builtin_amdgcn_wmma_i32_16x16x64_iu8(true, aF1, false, bF, acc[1][pt], false, false);
      }
    }
  }

  int mhalf = (lane >> 4) << 3;
  if (MODE == 0) {
    float mx = 0.f;
    const float* alpha = scal + SC_ALPHA + e * 256;
    const float* beta = scal + SC_BNB;
    for (int cot = 0; cot < 2; cot++) {
      int cbase = wave * 32 + cot * 16 + mhalf;
      for (int v = 0; v < 8; v++) {
        float al = alpha[cbase + v], bt = beta[cbase + v];
        for (int pt = 0; pt < 8; pt++)
          mx = fmaxf(mx, (float)acc[cot][pt][v] * al + bt);
      }
    }
    mx = fmaxf(mx, 0.f);
    for (int off = 16; off >= 1; off >>= 1) mx = fmaxf(mx, __shfl_xor(mx, off, 32));
    if (lane == 0) atomicMax(&amaxH[e], __float_as_uint(mx));
  } else if (MODE == 1) {
    float s2 = scal[SC_S2 + e];
    const float* alpha = scal + SC_ALPHA + e * 256;
    const float* beta = scal + SC_BNB;
    for (int cot = 0; cot < 2; cot++) {
      int cbase = wave * 32 + cot * 16 + mhalf;
      for (int pt = 0; pt < 8; pt++) {
        int posg = rowbase * 32 + pt * 16 + n;
        unsigned lo = 0, hi = 0;
        for (int v = 0; v < 8; v++) {
          float h = (float)acc[cot][pt][v] * alpha[cbase + v] + beta[cbase + v];
          h = fmaxf(h, 0.f);
          int q = min((int)rintf(h * s2), 255);
          if (v < 4) lo |= ((unsigned)q) << (8 * v);
          else       hi |= ((unsigned)q) << (8 * (v - 4));
        }
        unsigned* d = (unsigned*)(qa2 + ((size_t)b * 1024 + posg) * 256 + cbase);
        d[0] = lo;
        d[1] = hi;
      }
    }
  } else {
    float inv2 = scal[SC_INV2 + e];
    for (int cot = 0; cot < 2; cot++) {
      int cbase = wave * 32 + cot * 16 + mhalf;
      for (int pt = 0; pt < 8; pt++) {
        int posg = rowbase * 32 + pt * 16 + n;
        for (int v = 0; v < 8; v++)
          fout[((size_t)b * 256 + cbase + v) * 1024 + posg] =
              (float)acc[cot][pt][v] * inv2;
      }
    }
  }
}

__global__ void k_gn(const float* __restrict__ h, const float* __restrict__ x,
                     const float* __restrict__ gam, const float* __restrict__ bet,
                     float* __restrict__ out) {
  __shared__ float ssum[8], ssq[8];
  int b = blockIdx.x >> 2;
  int g = blockIdx.x & 3;
  int tid = threadIdx.x;
  size_t base = ((size_t)b * 256 + g * 64) * 1024;
  const float* hb = h + base;
  const float* xb = x + base;
  float* ob = out + base;
  float sum = 0.f, sq = 0.f;
  for (int i = tid; i < 64 * 1024; i += 256) {
    float v = hb[i];
    sum += v;
    sq += v * v;
  }
  for (int off = 16; off >= 1; off >>= 1) {
    sum += __shfl_xor(sum, off, 32);
    sq += __shfl_xor(sq, off, 32);
  }
  if ((tid & 31) == 0) { ssum[tid >> 5] = sum; ssq[tid >> 5] = sq; }
  __syncthreads();
  float ts = 0.f, tq = 0.f;
  for (int i = 0; i < 8; i++) { ts += ssum[i]; tq += ssq[i]; }
  float mean = ts * (1.f / 65536.f);
  float var = tq * (1.f / 65536.f) - mean * mean;
  float rstd = rsqrtf(fmaxf(var, 0.f) + EPS_);
  for (int i = tid; i < 64 * 1024; i += 256) {
    int c = g * 64 + (i >> 10);
    float o = (hb[i] - mean) * rstd * gam[c] + bet[c] + xb[i];
    ob[i] = fmaxf(o, 0.f);
  }
}

extern "C" void kernel_launch(void* const* d_in, const int* in_sizes, int n_in,
                              void* d_out, int out_size, void* d_ws, size_t ws_size,
                              hipStream_t stream) {
  (void)in_sizes; (void)n_in; (void)out_size; (void)ws_size;
  const float* x   = (const float*)d_in[0];
  const int* mask  = (const int*)d_in[1];
  const float* w1  = (const float*)d_in[2];
  const float* w2  = (const float*)d_in[3];
  const float* bng = (const float*)d_in[4];
  const float* bnb = (const float*)d_in[5];
  const float* bnm = (const float*)d_in[6];
  const float* bnv = (const float*)d_in[7];
  const float* gng = (const float*)d_in[8];
  const float* gnb = (const float*)d_in[9];

  char* ws = (char*)d_ws;
  float* scal = (float*)ws;
  unsigned* scalU = (unsigned*)ws;
  int* qw1 = (int*)(ws + 8192);
  int* qw2 = (int*)(ws + 8192 + 4 * QW_EXP_INTS * 3);
  unsigned char* qa1 = (unsigned char*)(ws + 8192 + 2 * 4 * QW_EXP_INTS * 3);
  unsigned char* qa2 = qa1 + 3ull * QA_EXP_BYTES;
  float* out2 = (float*)(qa2 + QA_EXP_BYTES);

  hipMemsetAsync(scal, 0, 8192, stream);
  k_amax<<<1024, 256, 0, stream>>>(x, 32 * 256 * 1024, 0, scalU + 0);
  k_amax<<<256, 256, 0, stream>>>(w1, 256 * 256 * 9, 1, scalU + 1);
  k_amax<<<256, 256, 0, stream>>>(w2, 256 * 256 * 9, 1, scalU + 2);
  k_derive1<<<1, 256, 0, stream>>>(scal, bng, bnb, bnm, bnv);
  k_pack_w<<<1728, 256, 0, stream>>>(w1, qw1, scal, SC_SW1);
  k_pack_w<<<1728, 256, 0, stream>>>(w2, qw2, scal, SC_SW2);
  k_quant_x<<<24576, 256, 0, stream>>>(x, qa1, scal);
  k_conv<0><<<768, 256, 0, stream>>>(qa1, qw1, scal, mask, nullptr, nullptr, scalU + 3);
  k_derive2<<<1, 32, 0, stream>>>(scal);
  k_conv<1><<<256, 256, 0, stream>>>(qa1, qw1, scal, mask, qa2, nullptr, nullptr);
  k_conv<2><<<256, 256, 0, stream>>>(qa2, qw2, scal, mask, nullptr, out2, nullptr);
  k_gn<<<128, 256, 0, stream>>>(out2, x, gng, gnb, (float*)d_out);
}
